// AttentionLayer_52682068853341
// MI455X (gfx1250) — matrix operand order repaired and checked by reading, not compiled, not run on hardware
//
#include <hip/hip_runtime.h>

// Problem constants (from reference)
#define B_   8
#define TE_  512
#define TD_  128
#define DE_  256
#define DQ_  256
#define A_   128

typedef __attribute__((ext_vector_type(16))) _Float16 v16h;
typedef __attribute__((ext_vector_type(4)))  _Float16 v4h;
typedef __attribute__((ext_vector_type(8)))  float    v8f;

__device__ __forceinline__ int lane_id() { return threadIdx.x & 31; }

__device__ __forceinline__ float fast_exp(float x) {
  return __builtin_amdgcn_exp2f(x * 1.4426950408889634f);
}

__device__ __forceinline__ float fast_tanh(float x) {
#if __has_builtin(__builtin_amdgcn_tanhf)
  return __builtin_amdgcn_tanhf(x);            // gfx1250 native v_tanh_f32
#else
  float e = __builtin_amdgcn_exp2f(x * 2.8853900817779268f); // exp(2x)
  return 1.0f - 2.0f * __builtin_amdgcn_rcpf(e + 1.0f);
#endif
}

// ---- WMMA fragment loaders (layouts per CDNA5 ISA 7.12.2, wave32) ----

// A fragment: 16x32 f16, rows r0..r0+15, cols k0..k0+31, from f32 row-major src[ld]
// lane L: M = L%16; half h=L/16; VGPR v(0..3): K = h*8 + 2v{,+1}; v(4..7): K = 16 + h*8 + 2(v-4){,+1}
__device__ __forceinline__ v16h load_a_f32(const float* __restrict__ src, int ld, int r0, int k0) {
  int l = lane_id();
  int h = l >> 4;
  const float* rp = src + (size_t)(r0 + (l & 15)) * ld + k0;
  v16h a;
#pragma unroll
  for (int v = 0; v < 8; ++v) {
    int k = ((v < 4) ? 0 : 16) + ((v & 3) << 1) + (h << 3);
    a[2 * v + 0] = (_Float16)rp[k];
    a[2 * v + 1] = (_Float16)rp[k + 1];
  }
  return a;
}

// Same A-fragment shape but sourced from f16 LDS (rows start at 0)
__device__ __forceinline__ v16h load_a_lds16(const _Float16* src, int ld, int k0) {
  int l = lane_id();
  int h = l >> 4;
  const _Float16* rp = src + (l & 15) * ld + k0;
  v16h a;
#pragma unroll
  for (int v = 0; v < 8; ++v) {
    int k = ((v < 4) ? 0 : 16) + ((v & 3) << 1) + (h << 3);
    a[2 * v + 0] = rp[k];
    a[2 * v + 1] = rp[k + 1];
  }
  return a;
}

// B fragment: 32x16 f16 from f16 LDS row-major [ld]: B[k][n] = src[(k0+k)*ld + n0+n]
// lane L: N = L%16; half h=L/16; VGPR v(0..3): K = h*8 + 2v{,+1}; v(4..7): K = 16 + h*8 + 2(v-4){,+1}
__device__ __forceinline__ v16h load_b_lds16(const _Float16* src, int ld, int k0, int n0) {
  int l = lane_id();
  const _Float16* cp = src + (size_t)(k0 + ((l >> 4) << 3)) * ld + n0 + (l & 15);
  v16h b;
#pragma unroll
  for (int v = 0; v < 8; ++v) {
    int kr = ((v < 4) ? 0 : 16) + ((v & 3) << 1);
    b[2 * v + 0] = cp[kr * ld];
    b[2 * v + 1] = cp[(kr + 1) * ld];
  }
  return b;
}

// C/D store: lane L: N = L%16; VGPR v -> M = v + 8*(L/16)
__device__ __forceinline__ void store_c_f32(float* __restrict__ dst, int ld, int r0, int n0, v8f c) {
  int l = lane_id();
  int n = l & 15, mh = (l >> 4) << 3;
#pragma unroll
  for (int v = 0; v < 8; ++v)
    dst[(size_t)(r0 + mh + v) * ld + n0 + n] = c[v];
}

// ---------------------------------------------------------------------------
// Kernel 1: projections.  Wh = values @ W_a  (4096x128, K=256)
//                         Us = query  @ U_a  (1024x128, K=256)
// Weights staged once per block into LDS as f16 (64 KB), 2 M-tiles per block.
// grid.x = 4096/32 + 1024/32 = 160 blocks of 256 threads (8 waves = 8 N-tiles)
// ---------------------------------------------------------------------------
__global__ __launch_bounds__(256) void attn_proj_kernel(
    const float* __restrict__ values, const float* __restrict__ query,
    const float* __restrict__ Wa, const float* __restrict__ Ua,
    float* __restrict__ Wh, float* __restrict__ Us) {
  __shared__ __align__(16) _Float16 sWt[DE_ * A_];   // [256 k][128 n] f16 = 64 KB

  int t = blockIdx.x;
  const float* src;
  const float* wm;
  float* dst;
  int r0;
  if (t < (B_ * TE_) / 32) {
    src = values; wm = Wa; dst = Wh; r0 = t * 32;
  } else {
    src = query;  wm = Ua; dst = Us; r0 = (t - (B_ * TE_) / 32) * 32;
  }

  // Coalesced stage of the 256x128 weight matrix, f32 -> f16 in LDS
  const float4* wm4 = (const float4*)wm;             // 32 float4 per k-row
#pragma unroll
  for (int j = 0; j < 32; ++j) {
    int idx = threadIdx.x + (j << 8);                // 0..8191
    int k = idx >> 5, n4 = (idx & 31) << 2;
    float4 v4 = wm4[idx];
    v4h h4;
    h4[0] = (_Float16)v4.x; h4[1] = (_Float16)v4.y;
    h4[2] = (_Float16)v4.z; h4[3] = (_Float16)v4.w;
    *(v4h*)(sWt + (size_t)k * A_ + n4) = h4;
  }
  __syncthreads();

  int n0 = (threadIdx.x >> 5) * 16;                  // wave id -> N tile of A_=128
#pragma unroll
  for (int mt = 0; mt < 2; ++mt) {
    v8f c = {};
#pragma unroll
    for (int k0 = 0; k0 < DE_; k0 += 32) {
      v16h a  = load_a_f32(src, DE_, r0 + mt * 16, k0);
      v16h bb = load_b_lds16(sWt, A_, k0, n0);
      c = __builtin_amdgcn_wmma_f32_16x16x32_f16(false, a, false, bb, (short)0, c, false, false);
    }
    store_c_f32(dst, A_, r0 + mt * 16, n0, c);
  }
}

// ---------------------------------------------------------------------------
// Kernel 2: fused scores -> softmax -> context.  One block per (b, 16 queries).
// 512 threads = 16 waves. The [B,TD,TE,A] act tensor is never materialized.
// grid.x = B_ * TD_/16 = 64
// ---------------------------------------------------------------------------
__global__ __launch_bounds__(512) void attn_fused_kernel(
    const float* __restrict__ values, const float* __restrict__ Wh,
    const float* __restrict__ Us, const float* __restrict__ Va,
    float* __restrict__ out_c, float* __restrict__ out_e) {
  __shared__ __align__(16) float    sUs[16 * A_];     // 8 KB
  __shared__ __align__(16) float    sV[A_];           // 0.5 KB
  __shared__ __align__(16) float    sS[16 * TE_];     // 32 KB scores; reused in phase 3
  __shared__ __align__(16) _Float16 sW[16 * TE_];     // 16 KB softmax weights (f16)

  int b  = blockIdx.x >> 3;
  int q0 = (blockIdx.x & 7) * 16;
  int w  = threadIdx.x >> 5;   // wave id 0..15
  int l  = threadIdx.x & 31;

  // stage Us tile + V into LDS
  for (int i = threadIdx.x; i < 16 * A_; i += 512)
    sUs[i] = Us[(size_t)(b * TD_ + q0 + (i >> 7)) * A_ + (i & (A_ - 1))];
  if (threadIdx.x < A_) sV[threadIdx.x] = Va[threadIdx.x];
  __syncthreads();

  // ---- Phase 1: scores[q][e] = sum_a V[a] * tanh(Wh[b,e,a] + Us[b,q,a]) ----
  {
    int q = w;                                   // one query per wave
    const float4* us4 = (const float4*)(sUs + q * A_);
    const float4* v4p = (const float4*)sV;
    const float* whb = Wh + (size_t)b * TE_ * A_;
#pragma unroll 1
    for (int j = 0; j < 16; ++j) {               // 16 encoder positions per lane
      int e = l + (j << 5);
      if (j < 15)                                 // gfx1250 global_prefetch_b8
        __builtin_prefetch(whb + (size_t)(e + 32) * A_, 0, 0);
      const float4* wh4 = (const float4*)(whb + (size_t)e * A_);
      float acc = 0.f;
#pragma unroll 8
      for (int a4 = 0; a4 < A_ / 4; ++a4) {
        float4 wv = wh4[a4];
        float4 uv = us4[a4];
        float4 vv = v4p[a4];
        acc += vv.x * fast_tanh(wv.x + uv.x);
        acc += vv.y * fast_tanh(wv.y + uv.y);
        acc += vv.z * fast_tanh(wv.z + uv.z);
        acc += vv.w * fast_tanh(wv.w + uv.w);
      }
      sS[q * TE_ + e] = acc;
    }
  }
  __syncthreads();

  // ---- Phase 2: softmax over TE per query (wave32 shuffle reductions) ----
  {
    int q = w;
    float mx = -3.0e38f;
#pragma unroll
    for (int j = 0; j < 16; ++j) mx = fmaxf(mx, sS[q * TE_ + l + (j << 5)]);
#pragma unroll
    for (int off = 16; off > 0; off >>= 1) mx = fmaxf(mx, __shfl_xor(mx, off, 32));
    float sum = 0.f;
#pragma unroll
    for (int j = 0; j < 16; ++j) sum += fast_exp(sS[q * TE_ + l + (j << 5)] - mx);
#pragma unroll
    for (int off = 16; off > 0; off >>= 1) sum += __shfl_xor(sum, off, 32);
    float inv = 1.0f / sum;
    float* oe = out_e + (size_t)(b * TD_ + q0 + q) * TE_;
#pragma unroll
    for (int j = 0; j < 16; ++j) {
      int e = l + (j << 5);
      float wt = fast_exp(sS[q * TE_ + e] - mx) * inv;
      oe[e] = wt;                                // output e (f32)
      sW[q * TE_ + e] = (_Float16)wt;            // keep f16 copy for WMMA
    }
  }
  __syncthreads();                               // sW complete; sS now dead

  // ---- Phase 3: context c[q,:] = weights @ values[b]  (16 x 512 @ 512 x 256) ----
  // values k-tiles staged into LDS (f16, double-buffered in the dead sS space)
  // and shared by all 16 waves; one barrier per k-step (ping-pong protects WAR).
  {
    int n0 = w << 4;                             // one 16-wide N tile per wave
    const float4* vb4 = (const float4*)(values + (size_t)b * TE_ * DE_);
    _Float16* tiles = (_Float16*)sS;             // 2 buffers x (32*256) halfs = 32 KB
    v8f c = {};
#pragma unroll 1
    for (int kk = 0; kk < TE_; kk += 32) {
      _Float16* tb = tiles + ((kk >> 5) & 1) * (32 * DE_);
      // cooperative stage: 32 x 256 f32 -> f16 (coalesced b128 reads, b64 LDS writes)
#pragma unroll
      for (int j = 0; j < 4; ++j) {
        int idx = threadIdx.x + (j << 9);        // 0..2047 float4 chunks
        int e = idx >> 6, d4 = idx & 63;
        float4 v4 = vb4[(size_t)(kk + e) * (DE_ / 4) + d4];
        v4h h4;
        h4[0] = (_Float16)v4.x; h4[1] = (_Float16)v4.y;
        h4[2] = (_Float16)v4.z; h4[3] = (_Float16)v4.w;
        *(v4h*)(tb + (size_t)e * DE_ + (d4 << 2)) = h4;
      }
      __syncthreads();
      v16h a  = load_a_lds16(sW, TE_, kk);
      v16h bb = load_b_lds16(tb, DE_, 0, n0);
      c = __builtin_amdgcn_wmma_f32_16x16x32_f16(false, a, false, bb, (short)0, c, false, false);
    }
    int n = l & 15, mh = (l >> 4) << 3;
#pragma unroll
    for (int v = 0; v < 8; ++v)
      out_c[(size_t)(b * TD_ + q0 + mh + v) * DE_ + n0 + n] = c[v];
  }
}

extern "C" void kernel_launch(void* const* d_in, const int* in_sizes, int n_in,
                              void* d_out, int out_size, void* d_ws, size_t ws_size,
                              hipStream_t stream) {
  const float* values = (const float*)d_in[0];  // [8,512,256]
  const float* query  = (const float*)d_in[1];  // [8,128,256]
  const float* W_a    = (const float*)d_in[2];  // [256,128]
  const float* U_a    = (const float*)d_in[3];  // [256,128]
  const float* V_a    = (const float*)d_in[4];  // [128,1]
  (void)in_sizes; (void)n_in; (void)out_size; (void)ws_size;

  float* out_c = (float*)d_out;                    // [8,128,256]
  float* out_e = out_c + (size_t)B_ * TD_ * DE_;   // [8,128,512]

  float* Wh = (float*)d_ws;                        // [8*512,128] = 2 MB
  float* Us = Wh + (size_t)B_ * TE_ * A_;          // [8*128,128] = 0.5 MB

  attn_proj_kernel<<<dim3((B_ * TE_) / 32 + (B_ * TD_) / 32), dim3(256), 0, stream>>>(
      values, query, W_a, U_a, Wh, Us);
  attn_fused_kernel<<<dim3(B_ * (TD_ / 16)), dim3(512), 0, stream>>>(
      values, Wh, Us, V_a, out_c, out_e);
}
